// Attention_9629316678041
// MI455X (gfx1250) — matrix operand order repaired and checked by reading, not compiled, not run on hardware
//
#include <hip/hip_runtime.h>

// Problem dims (fixed by the reference)
#define B_    4
#define N_    2048
#define DIM_  1024
#define H_    16
#define DH_   64
#define ROWS_ (B_ * N_)      // 8192
#define SCALE_ 0.125f        // 64^-0.5

typedef float  v8f   __attribute__((ext_vector_type(8)));
typedef __bf16 v16bf __attribute__((ext_vector_type(16)));
typedef unsigned int u32x4 __attribute__((ext_vector_type(4)));
typedef int i32x4 __attribute__((ext_vector_type(4)));
typedef int i32x8 __attribute__((ext_vector_type(8)));

#if defined(__HIP_DEVICE_COMPILE__) && __has_builtin(__builtin_amdgcn_tensor_load_to_lds)
#define USE_TDM 1
#else
#define USE_TDM 0
#endif

__device__ __forceinline__ __bf16 f2bf(float f) {
  unsigned u = __builtin_bit_cast(unsigned, f);
  u += 0x7FFFu + ((u >> 16) & 1u);                 // round-to-nearest-even
  unsigned short h = (unsigned short)(u >> 16);
  return __builtin_bit_cast(__bf16, h);
}

__device__ __forceinline__ v8f wmma_bf16(v16bf a, v16bf b, v8f c) {
  // D = A(16x32) * B(32x16) + C(16x16 f32)
  return __builtin_amdgcn_wmma_f32_16x16x32_bf16(false, a, false, b, (short)0, c,
                                                 false, false);
}

// A-fragment 16x32 (M x K), 16-bit, row-major source, leading dim ld.
// Element e -> K = k0 + 16*half + e
__device__ __forceinline__ v16bf load_a_frag(const __bf16* __restrict__ A, int ld,
                                             int row, int k0, int half) {
  const __bf16* p = A + (size_t)row * ld + k0 + half * 16;
  v16bf a;
#pragma unroll
  for (int e = 0; e < 8; ++e) a[e] = p[e];
#pragma unroll
  for (int e = 0; e < 8; ++e) a[8 + e] = p[8 + e];
  return a;
}

// B-fragment 32x16 (K x N): element e -> K = 16*half + e, N = lane%16.
__device__ __forceinline__ v16bf load_b_contig(const __bf16* __restrict__ p) {
  v16bf b;
#pragma unroll
  for (int e = 0; e < 16; ++e) b[e] = p[e];
  return b;
}

__device__ __forceinline__ v16bf load_b_strided(const __bf16* __restrict__ p, int stride) {
  v16bf b;
#pragma unroll
  for (int e = 0; e < 16; ++e) b[e] = p[(size_t)e * stride];
  return b;
}

#if USE_TDM
// TDM: DMA a 64-row x 32-col bf16 tile from A[8192][1024] (row-major) into LDS.
// D# per CDNA5 ISA: group0 = {flags, lds_addr,
// global_addr[56:0], type=2}; group1 = {data_size=2B, tensor_dim0=1024,
// tensor_dim1=8192, tile_dim0=32, tile_dim1=64, dim0_stride=1024}.
__device__ __forceinline__ void tdm_load_a64x32(const __bf16* gsrc, unsigned lds_off) {
  unsigned long long ga = (unsigned long long)(uintptr_t)gsrc;
  u32x4 g0;
  g0[0] = 1u;                                            // count=1, user mode
  g0[1] = lds_off;                                       // lds_addr (bytes)
  g0[2] = (unsigned)ga;                                  // global_addr[31:0]
  g0[3] = (unsigned)((ga >> 32) & 0x1FFFFFFu) | (2u << 30); // addr[56:32] | type=2
  i32x8 g1;
  g1[0] = 0x00010000;   // wg_mask=0 | data_size=1 (2 bytes)
  g1[1] = 0x04000000;   // atomic_barrier=0 | tensor_dim0(=1024).lo16 << 16
  g1[2] = 0x20000000;   // tensor_dim0.hi16=0 | tensor_dim1(=8192).lo16 << 16
  g1[3] = 0x00200000;   // tensor_dim1.hi16=0 | tile_dim0(=32) << 16
  g1[4] = 0x00000040;   // tile_dim1=64 | tile_dim2=0
  g1[5] = 0x00000400;   // tensor_dim0_stride = 1024 (low 32)
  g1[6] = 0;            // stride0.hi16=0 | dim1_stride.lo16=0
  g1[7] = 0;
  i32x4 z4 = {0, 0, 0, 0};
#if defined(__clang_major__) && __clang_major__ >= 23
  i32x8 z8 = {0, 0, 0, 0, 0, 0, 0, 0};
  __builtin_amdgcn_tensor_load_to_lds(g0, g1, z4, z4, z8, 0);
#else
  __builtin_amdgcn_tensor_load_to_lds(g0, g1, z4, z4, 0);
#endif
}
#endif // USE_TDM

// ---------------------------------------------------------------------------
// Shared GEMM core: block = 128 threads (4 waves). Block tile M=64, N=128;
// each wave computes 64x32 (4x2 WMMA frags, 8 WMMAs per K-step). The 64x32
// A-tile is staged into LDS (TDM double-buffered when available) and shared
// by all 4 waves; B fragments stream from global (L2-resident weights).
// ---------------------------------------------------------------------------
__device__ __forceinline__ void gemm_core(const __bf16* __restrict__ A,
                                          const __bf16* __restrict__ W,
                                          int m0, int n0, int lane, int wave,
                                          __bf16 (*Abuf)[64 * 32],
                                          v8f acc[4][2]) {
  const int half = lane >> 4;
  const int lr   = lane & 15;
  const int K    = DIM_;
#if USE_TDM
  if (wave == 0)
    tdm_load_a64x32(A + (size_t)m0 * K, (unsigned)(uintptr_t)&Abuf[0][0]);
#endif
  int cur = 0;
  for (int k0 = 0; k0 < K; k0 += 32) {
#if USE_TDM
    if (wave == 0) {
      if (k0 + 32 < K) {
        tdm_load_a64x32(A + (size_t)m0 * K + (k0 + 32),
                        (unsigned)(uintptr_t)&Abuf[cur ^ 1][0]);
        __builtin_amdgcn_s_wait_tensorcnt(1);   // oldest (current tile) done
      } else {
        __builtin_amdgcn_s_wait_tensorcnt(0);   // last tile done
      }
    }
    __syncthreads();
#else
    __syncthreads();
    for (int idx = threadIdx.x; idx < 64 * 32; idx += 128) {
      int r = idx >> 5, c = idx & 31;
      Abuf[0][idx] = A[(size_t)(m0 + r) * K + k0 + c];
    }
    __syncthreads();
#endif
    __builtin_prefetch(W + (size_t)(n0 + lr) * K + k0 + 128, 0, 1);
    v16bf b0 = load_b_contig(W + (size_t)(n0 + lr)      * K + k0 + 16 * half);
    v16bf b1 = load_b_contig(W + (size_t)(n0 + 16 + lr) * K + k0 + 16 * half);
#pragma unroll
    for (int i = 0; i < 4; ++i) {
      v16bf a = load_a_frag(&Abuf[cur][0], 32, 16 * i + lr, 0, half);
      acc[i][0] = wmma_bf16(a, b0, acc[i][0]);
      acc[i][1] = wmma_bf16(a, b1, acc[i][1]);
    }
#if USE_TDM
    __syncthreads();          // all waves done reading before TDM reuses buffer
    cur ^= 1;
#endif
  }
}

// ---------------------------------------------------------------------------
// Kernel 1: LayerNorm (fp32 in) -> bf16 normalized activations. 1 block / row.
// ---------------------------------------------------------------------------
__global__ __launch_bounds__(256) void ln_kernel(const float* __restrict__ x,
                                                 const float* __restrict__ g,
                                                 const float* __restrict__ be,
                                                 __bf16* __restrict__ xn) {
  __shared__ float s1[256], s2[256];
  const int row = blockIdx.x;
  const int tid = threadIdx.x;
  const float* xr = x + (size_t)row * DIM_;
  float4 v = ((const float4*)xr)[tid];
  s1[tid] = v.x + v.y + v.z + v.w;
  s2[tid] = v.x * v.x + v.y * v.y + v.z * v.z + v.w * v.w;
  __syncthreads();
  for (int off = 128; off > 0; off >>= 1) {
    if (tid < off) { s1[tid] += s1[tid + off]; s2[tid] += s2[tid + off]; }
    __syncthreads();
  }
  float mu   = s1[0] * (1.0f / DIM_);
  float var  = s2[0] * (1.0f / DIM_) - mu * mu;
  float rstd = rsqrtf(var + 1e-5f);
  int i = tid * 4;
  __bf16* o = xn + (size_t)row * DIM_ + i;
  o[0] = f2bf((v.x - mu) * rstd * g[i + 0] + be[i + 0]);
  o[1] = f2bf((v.y - mu) * rstd * g[i + 1] + be[i + 1]);
  o[2] = f2bf((v.z - mu) * rstd * g[i + 2] + be[i + 2]);
  o[3] = f2bf((v.w - mu) * rstd * g[i + 3] + be[i + 3]);
}

// ---------------------------------------------------------------------------
// Kernel 2: fp32 -> bf16 weight downconvert
// ---------------------------------------------------------------------------
__global__ __launch_bounds__(256) void cvt_kernel(const float* __restrict__ src,
                                                  __bf16* __restrict__ dst, int n) {
  int i = blockIdx.x * 256 + threadIdx.x;
  if (i < n) dst[i] = f2bf(src[i]);
}

// ---------------------------------------------------------------------------
// Kernel 3: QKV GEMM. A=[8192,1024] bf16, W=[3072,1024] bf16 ([out,in]).
// Epilogue scatters to per-head q/k/v [b, h, n, 64] bf16.
// ---------------------------------------------------------------------------
__global__ __launch_bounds__(128) void qkv_gemm(const __bf16* __restrict__ A,
                                                const __bf16* __restrict__ W,
                                                __bf16* __restrict__ Qo,
                                                __bf16* __restrict__ Ko,
                                                __bf16* __restrict__ Vo) {
  __shared__ __bf16 Abuf[2][64 * 32];
  const int lane = threadIdx.x & 31;
  const int wave = threadIdx.x >> 5;
  const int half = lane >> 4;
  const int lr   = lane & 15;
  const int m0   = blockIdx.y * 64;
  const int n0   = blockIdx.x * 128 + wave * 32;

  v8f acc[4][2] = {};
  gemm_core(A, W, m0, n0, lane, wave, Abuf, acc);

#pragma unroll
  for (int i = 0; i < 4; ++i) {
#pragma unroll
    for (int j = 0; j < 2; ++j) {
      const int ncol  = n0 + 16 * j;             // uniform across lanes
      const int which = ncol >> 10;              // 0=q, 1=k, 2=v
      __bf16* dst = (which == 0) ? Qo : (which == 1) ? Ko : Vo;
      const int r    = (ncol & 1023) + lr;
      const int head = r >> 6;
      const int d    = r & 63;
#pragma unroll
      for (int v = 0; v < 8; ++v) {
        const int m   = m0 + 16 * i + v + 8 * half;
        const int bi  = m >> 11;                 // m / 2048
        const int row = m & 2047;
        dst[(((size_t)(bi * H_ + head)) * N_ + row) * DH_ + d] = f2bf(acc[i][j][v]);
      }
    }
  }
}

// ---------------------------------------------------------------------------
// Kernel 4: flash attention. One wave per 16-row Q tile; 128 threads = 4 waves.
// Per 32-key block: 4 score WMMAs, online softmax, P restaged via LDS to
// A-fragment layout, 4 P*V WMMAs. Output -> attn [b, n, h*64+d] bf16.
// ---------------------------------------------------------------------------
__global__ __launch_bounds__(128) void attn_kernel(const __bf16* __restrict__ Q,
                                                   const __bf16* __restrict__ Km,
                                                   const __bf16* __restrict__ Vm,
                                                   __bf16* __restrict__ attn) {
  __shared__ __bf16 ldsP[4][16][32];             // wave-private 16x32 P tile
  const int lane = threadIdx.x & 31;
  const int wave = threadIdx.x >> 5;
  const int half = lane >> 4;
  const int lr   = lane & 15;

  const int t  = blockIdx.x * 4 + wave;          // 0..8191
  const int bi = t >> 11;
  const int h  = (t & 2047) >> 7;
  const int q0 = (t & 127) * 16;

  const size_t headoff = ((size_t)(bi * H_ + h)) * N_ * DH_;
  const __bf16* Qp = Q  + headoff;
  const __bf16* Kp = Km + headoff;
  const __bf16* Vp = Vm + headoff;

  v16bf qf0 = load_a_frag(Qp, DH_, q0 + lr, 0,  half);
  v16bf qf1 = load_a_frag(Qp, DH_, q0 + lr, 32, half);

  float mrow[8], lrow[8];
  v8f o[4] = {};
#pragma unroll
  for (int v = 0; v < 8; ++v) { mrow[v] = -1e30f; lrow[v] = 0.0f; }

  for (int j0 = 0; j0 < N_; j0 += 32) {
    v8f s[2] = {};
#pragma unroll
    for (int tt = 0; tt < 2; ++tt) {
      const __bf16* kp = Kp + (size_t)(j0 + 16 * tt + lr) * DH_ + 16 * half;
      s[tt] = wmma_bf16(qf0, load_b_contig(kp),      s[tt]);
      s[tt] = wmma_bf16(qf1, load_b_contig(kp + 32), s[tt]);
    }

    float mnew[8], alpha[8];
#pragma unroll
    for (int v = 0; v < 8; ++v) {
      float s0 = s[0][v] * SCALE_;
      float s1 = s[1][v] * SCALE_;
      s[0][v] = s0; s[1][v] = s1;
      float mx = fmaxf(s0, s1);
      mx = fmaxf(mx, __shfl_xor(mx, 1, 16));
      mx = fmaxf(mx, __shfl_xor(mx, 2, 16));
      mx = fmaxf(mx, __shfl_xor(mx, 4, 16));
      mx = fmaxf(mx, __shfl_xor(mx, 8, 16));
      mnew[v]  = fmaxf(mrow[v], mx);
      alpha[v] = __expf(mrow[v] - mnew[v]);
      mrow[v]  = mnew[v];
    }
#pragma unroll
    for (int v = 0; v < 8; ++v) {
      float p0 = __expf(s[0][v] - mnew[v]);
      float p1 = __expf(s[1][v] - mnew[v]);
      float rs = p0 + p1;
      rs += __shfl_xor(rs, 1, 16);
      rs += __shfl_xor(rs, 2, 16);
      rs += __shfl_xor(rs, 4, 16);
      rs += __shfl_xor(rs, 8, 16);
      lrow[v] = lrow[v] * alpha[v] + rs;
      ldsP[wave][v + 8 * half][lr]      = f2bf(p0);
      ldsP[wave][v + 8 * half][16 + lr] = f2bf(p1);
      o[0][v] *= alpha[v]; o[1][v] *= alpha[v];
      o[2][v] *= alpha[v]; o[3][v] *= alpha[v];
    }

    v16bf pf = load_a_frag(&ldsP[wave][0][0], 32, lr, 0, half);
#pragma unroll
    for (int c = 0; c < 4; ++c) {
      v16bf vb = load_b_strided(Vp + (size_t)(j0 + 16 * half) * DH_ + 16 * c + lr, DH_);
      o[c] = wmma_bf16(pf, vb, o[c]);
    }
  }

#pragma unroll
  for (int v = 0; v < 8; ++v) {
    float inv = 1.0f / lrow[v];
    const int row = q0 + v + 8 * half;
    __bf16* dst = attn + ((size_t)bi * N_ + row) * (H_ * DH_) + h * DH_;
#pragma unroll
    for (int c = 0; c < 4; ++c)
      dst[c * 16 + lr] = f2bf(o[c][v] * inv);
  }
}

// ---------------------------------------------------------------------------
// Kernel 5: out projection. A = attn [8192,1024] bf16, W = [1024,1024] bf16,
// C = f32 out [8192,1024] + bias.
// ---------------------------------------------------------------------------
__global__ __launch_bounds__(128) void out_gemm(const __bf16* __restrict__ A,
                                                const __bf16* __restrict__ W,
                                                const float* __restrict__ bias,
                                                float* __restrict__ C) {
  __shared__ __bf16 Abuf[2][64 * 32];
  const int lane = threadIdx.x & 31;
  const int wave = threadIdx.x >> 5;
  const int half = lane >> 4;
  const int lr   = lane & 15;
  const int m0   = blockIdx.y * 64;
  const int n0   = blockIdx.x * 128 + wave * 32;

  v8f acc[4][2] = {};
  gemm_core(A, W, m0, n0, lane, wave, Abuf, acc);

#pragma unroll
  for (int i = 0; i < 4; ++i) {
#pragma unroll
    for (int j = 0; j < 2; ++j) {
      const int n = n0 + 16 * j + lr;
      const float bv = bias[n];
#pragma unroll
      for (int v = 0; v < 8; ++v) {
        const int m = m0 + 16 * i + v + 8 * half;
        C[(size_t)m * DIM_ + n] = acc[i][j][v] + bv;
      }
    }
  }
}

// ---------------------------------------------------------------------------
extern "C" void kernel_launch(void* const* d_in, const int* in_sizes, int n_in,
                              void* d_out, int out_size, void* d_ws, size_t ws_size,
                              hipStream_t stream) {
  (void)in_sizes; (void)n_in; (void)out_size; (void)ws_size;
  const float* x     = (const float*)d_in[0];
  const float* gamma = (const float*)d_in[1];
  const float* beta  = (const float*)d_in[2];
  const float* w_qkv = (const float*)d_in[3];
  const float* w_out = (const float*)d_in[4];
  const float* b_out = (const float*)d_in[5];
  float* out = (float*)d_out;

  char* ws = (char*)d_ws;
  const size_t ACT = (size_t)ROWS_ * DIM_ * sizeof(unsigned short);   // 16 MB
  __bf16* xn  = (__bf16*)ws;                 ws += ACT;
  __bf16* wqb = (__bf16*)ws;                 ws += (size_t)3 * DIM_ * DIM_ * 2;
  __bf16* wob = (__bf16*)ws;                 ws += (size_t)DIM_ * DIM_ * 2;
  __bf16* qb  = (__bf16*)ws;                 ws += ACT;
  __bf16* kb  = (__bf16*)ws;                 ws += ACT;
  __bf16* vb  = (__bf16*)ws;                 ws += ACT;
  __bf16* ab  = (__bf16*)ws;                 ws += ACT;

  // 1) LayerNorm -> bf16
  ln_kernel<<<dim3(ROWS_), dim3(256), 0, stream>>>(x, gamma, beta, xn);

  // 2) Weight downconverts
  {
    int n1 = 3 * DIM_ * DIM_;
    cvt_kernel<<<dim3((n1 + 255) / 256), dim3(256), 0, stream>>>(w_qkv, wqb, n1);
    int n2 = DIM_ * DIM_;
    cvt_kernel<<<dim3((n2 + 255) / 256), dim3(256), 0, stream>>>(w_out, wob, n2);
  }

  // 3) QKV GEMM: M=8192, N=3072 -> grid (3072/128, 8192/64)
  qkv_gemm<<<dim3(24, 128), dim3(128), 0, stream>>>(xn, wqb, qb, kb, vb);

  // 4) Flash attention: 8192 q-tiles / 4 waves per block
  attn_kernel<<<dim3(2048), dim3(128), 0, stream>>>(qb, kb, vb, ab);

  // 5) Out projection: N=1024 -> grid (1024/128, 8192/64)
  out_gemm<<<dim3(8, 128), dim3(128), 0, stream>>>(ab, wob, b_out, out);
}
